// MessagePassing_58007828300078
// MI455X (gfx1250) — matrix operand order repaired and checked by reading, not compiled, not run on hardware
//
#include <hip/hip_runtime.h>
#include <math.h>

// ---------- constants (match reference shapes) ----------
#define LMAX   6
#define SH_DIM 49          // (LMAX+1)^2
#define NUM_INV 7
#define H      64
#define R      64
#define C3H    192         // 3*H

typedef __attribute__((ext_vector_type(16))) __bf16    v16bf;
typedef __attribute__((ext_vector_type(8)))  float     v8f;
typedef __attribute__((ext_vector_type(4)))  float     f4v;
typedef __attribute__((ext_vector_type(2)))  float     f2v;
typedef __attribute__((ext_vector_type(4)))  __bf16    bf4v;
typedef __attribute__((ext_vector_type(2)))  __bf16    bf2v;
typedef __attribute__((ext_vector_type(4)))  unsigned  u4v;

union ABf { v16bf v; bf4v h4[4]; unsigned u[8]; u4v q[2]; };

// packed hardware f32x2 -> bf16x2 (v_cvt_pk_bf16_f32)
__device__ __forceinline__ unsigned bfpack(float lo, float hi) {
  f2v f; f.x = lo; f.y = hi;
  union { bf2v b; unsigned u; } r;
  r.b = __builtin_convertvector(f, bf2v);
  return r.u;
}

// convert 8 contiguous floats (16B-aligned) into 2 packed bf4v chunks
__device__ __forceinline__ void pack8(const float* src, bf4v* dst) {
  f4v a = *(const f4v*)src;
  f4v b = *(const f4v*)(src + 4);
  dst[0] = __builtin_convertvector(a, bf4v);
  dst[1] = __builtin_convertvector(b, bf4v);
}

__device__ __forceinline__ v8f wmma_bf16(const ABf& a, const ABf& b, v8f c) {
  return __builtin_amdgcn_wmma_f32_16x16x32_bf16(false, a.v, false, b.v,
                                                 (short)0, c, false, false);
}

// fast gate: tanh(silu(t)/0.6) with v_rcp_f32 / v_exp_f32 (no IEEE div chains)
__device__ __forceinline__ float sgate(float t) {
  float sig = __builtin_amdgcn_rcpf(1.0f + __expf(-t));   // sigmoid(t)
  float si  = t * sig * (1.0f / 0.6f);                    // scaled silu
  // tanh(x) = 1 - 2/(exp(2x)+1)
  return 1.0f - 2.0f * __builtin_amdgcn_rcpf(__expf(2.0f * si) + 1.0f);
}

__device__ __forceinline__ float ssilu(float t) {         // silu(t)/0.6
  return t * __builtin_amdgcn_rcpf(1.0f + __expf(-t)) * (1.0f / 0.6f);
}

// ---------- real spherical harmonics up to LMAX ----------
__device__ __forceinline__ void sph_full(float ux, float uy, float uz, float* out) {
  float invn = rsqrtf(ux*ux + uy*uy + uz*uz + 1e-12f);
  float x = ux*invn, y = uy*invn, z = uz*invn;
  float st = sqrtf(fmaxf(x*x + y*y, 0.0f));
  float phi = atan2f(y, x);
  float cp = cosf(phi), sphi = sinf(phi);
  const float S4PI = 3.5449077018110318f;
  const float SQ2  = 1.4142135623730951f;
  float pmm = 0.28209479177387814f;
  float cm = 1.0f, sm = 0.0f;
  #pragma unroll
  for (int m = 0; m <= LMAX; ++m) {
    if (m > 0) {
      pmm = -sqrtf((2.0f*m + 1.0f) / (2.0f*m)) * st * pmm;
      float c2 = cm*cp - sm*sphi;
      float s2 = sm*cp + cm*sphi;
      cm = c2; sm = s2;
    }
    float pl = pmm, plm1 = 0.0f;
    for (int l = m; l <= LMAX; ++l) {
      if (l == m + 1) { plm1 = pl; pl = sqrtf(2.0f*m + 3.0f) * z * pmm; }
      else if (l > m + 1) {
        float fl = (float)l;
        float a  = sqrtf((4.0f*fl*fl - 1.0f) / (fl*fl - (float)(m*m)));
        float bq = sqrtf(((fl-1.0f)*(fl-1.0f) - (float)(m*m)) /
                         (4.0f*(fl-1.0f)*(fl-1.0f) - 1.0f));
        float pn = a * (z*pl - bq*plm1);
        plm1 = pl; pl = pn;
      }
      int base = l*l + l;
      if (m == 0) out[base] = S4PI * pl;
      else { out[base - m] = S4PI*SQ2*pl*sm; out[base + m] = S4PI*SQ2*pl*cm; }
    }
  }
}

// incremental SH * gathered ref-SH dot, accumulated per l (no 49-elem array)
__device__ __forceinline__ void sph_dot7(float ux, float uy, float uz,
                                         const float* rs, float* accl) {
  float invn = rsqrtf(ux*ux + uy*uy + uz*uz + 1e-12f);
  float x = ux*invn, y = uy*invn, z = uz*invn;
  float st = sqrtf(fmaxf(x*x + y*y, 0.0f));
  float phi = atan2f(y, x);
  float cp = cosf(phi), sphi = sinf(phi);
  const float S4PI = 3.5449077018110318f;
  const float SQ2  = 1.4142135623730951f;
  float pmm = 0.28209479177387814f;
  float cm = 1.0f, sm = 0.0f;
  #pragma unroll
  for (int m = 0; m <= LMAX; ++m) {
    if (m > 0) {
      pmm = -sqrtf((2.0f*m + 1.0f) / (2.0f*m)) * st * pmm;
      float c2 = cm*cp - sm*sphi;
      float s2 = sm*cp + cm*sphi;
      cm = c2; sm = s2;
    }
    float pl = pmm, plm1 = 0.0f;
    for (int l = m; l <= LMAX; ++l) {
      if (l == m + 1) { plm1 = pl; pl = sqrtf(2.0f*m + 3.0f) * z * pmm; }
      else if (l > m + 1) {
        float fl = (float)l;
        float a  = sqrtf((4.0f*fl*fl - 1.0f) / (fl*fl - (float)(m*m)));
        float bq = sqrtf(((fl-1.0f)*(fl-1.0f) - (float)(m*m)) /
                         (4.0f*(fl-1.0f)*(fl-1.0f) - 1.0f));
        float pn = a * (z*pl - bq*plm1);
        plm1 = pl; pl = pn;
      }
      int base = l*l + l;
      if (m == 0) accl[l] += S4PI * pl * rs[base];
      else        accl[l] += S4PI*SQ2*pl * (sm*rs[base - m] + cm*rs[base + m]);
    }
  }
}

// ---------- kernel 1: pack weights into WMMA B-fragment order (bf16 pairs) ----
// Layout: word index = frag*256 + lane*8 + v  (each lane's 8 dwords contiguous
// -> two b128 loads per fragment in the consumer kernels).
// B-frag lane mapping (16-bit, KxN=32x16): N=lane%16, h=lane/16; dword v holds
// K=k,k+1 with k = 8h+2v (v<4) or 16+8h+2(v-4) (v>=4).
__global__ void pack_weights(const float* __restrict__ We,
                             const float* __restrict__ W1,
                             const float* __restrict__ W2,
                             unsigned* pWe, unsigned* pW1, unsigned* pW2) {
  int t = blockIdx.x * blockDim.x + threadIdx.x;
  if (t < 6144) {                                // W_edge 64x192: 24 frags
    int frag = t >> 8, lane = (t >> 3) & 31, v = t & 7;
    int nt = frag >> 1, kk = frag & 1;
    int col = nt*16 + (lane & 15);
    int K   = kk*32 + ((lane >> 4) * 8) + ((v >> 2) * 8) + 2*v;
    pWe[t] = bfpack(We[K*C3H + col], We[(K+1)*C3H + col]);
  } else if (t < 7168) {                         // W_x1 64x32: 4 frags
    int s = t - 6144;
    int frag = s >> 8, lane = (s >> 3) & 31, v = s & 7;
    int nt = frag >> 1, kk = frag & 1;
    int col = nt*16 + (lane & 15);
    int K   = kk*32 + ((lane >> 4) * 8) + ((v >> 2) * 8) + 2*v;
    pW1[s] = bfpack(W1[K*32 + col], W1[(K+1)*32 + col]);
  } else if (t < 10240) {                        // W_x2 32x192: 12 frags
    int s = t - 7168;
    int nt = s >> 8, lane = (s >> 3) & 31, v = s & 7;
    int col = nt*16 + (lane & 15);
    int K   = ((lane >> 4) * 8) + ((v >> 2) * 8) + 2*v;
    pW2[s] = bfpack(W2[K*C3H + col], W2[(K+1)*C3H + col]);
  }
}

__device__ __forceinline__ void load_bfrag(const unsigned* base, int frag, int lane, ABf& b) {
  const u4v* p = (const u4v*)(base + frag*256 + lane*8);
  b.q[0] = p[0];
  b.q[1] = p[1];
}

// ---------- kernel 2: scatter edge_udiff by destination node ----------
__global__ void edge_ref_scatter(const float* __restrict__ udiff,
                                 const int* __restrict__ eidx, int E,
                                 float* racc, float* cnt) {
  int e = blockIdx.x * blockDim.x + threadIdx.x;
  if (e >= E) return;
  int i = eidx[E + e];
  unsafeAtomicAdd(&racc[i*3 + 0], udiff[e*3 + 0]);
  unsafeAtomicAdd(&racc[i*3 + 1], udiff[e*3 + 1]);
  unsafeAtomicAdd(&racc[i*3 + 2], udiff[e*3 + 2]);
  unsafeAtomicAdd(&cnt[i], 1.0f);
}

// ---------- kernel 3: normalize ref_vec, compute per-node ref SH ----------
__global__ void node_ref(float* rvec, const float* __restrict__ cnt,
                         float* rsh, int N) {
  int n = blockIdx.x * blockDim.x + threadIdx.x;
  if (n >= N) return;
  float c = fmaxf(cnt[n], 1.0f);
  float vx = rvec[n*3+0] / c, vy = rvec[n*3+1] / c, vz = rvec[n*3+2] / c;
  float nr = sqrtf(vx*vx + vy*vy + vz*vz + 1e-9f);
  float rx = vx / nr, ry = vy / nr, rz = vz / nr;
  if (nr < 5e-5f) { rx = 1.0f; ry = 0.0f; rz = 0.0f; }
  rvec[n*3+0] = rx; rvec[n*3+1] = ry; rvec[n*3+2] = rz;
  float sh[SH_DIM];
  sph_full(rx, ry, rz, sh);
  #pragma unroll
  for (int k = 0; k < SH_DIM; ++k) rsh[n*SH_DIM + k] = sh[k];
}

// ---------- kernel 4: scalar_proj = silu(ns@W1+b1)@W2+b2 via WMMA ----------
__global__ void __launch_bounds__(128) scalar_proj_k(
    const float* __restrict__ ns, const unsigned* __restrict__ pW1,
    const unsigned* __restrict__ pW2,
    const float* __restrict__ b1, const float* __restrict__ b2,
    float* __restrict__ sp, int N) {
  __shared__ float hl[4][16][36];                 // stride 36 -> 16B aligned rows
  int lane = threadIdx.x & 31, w = threadIdx.x >> 5;
  int half = lane >> 4, lm = lane & 15;
  int n0 = (blockIdx.x*4 + w) * 16;

  int row = n0 + lm; if (row >= N) row = N - 1;
  ABf a0, a1;                                     // A frags 16x32 (K 0..31 / 32..63)
  pack8(&ns[row*H +      half*8], &a0.h4[0]);
  pack8(&ns[row*H + 16 + half*8], &a0.h4[2]);
  pack8(&ns[row*H + 32 + half*8], &a1.h4[0]);
  pack8(&ns[row*H + 48 + half*8], &a1.h4[2]);

  #pragma unroll
  for (int nt = 0; nt < 2; ++nt) {
    v8f c = {};
    ABf b;
    load_bfrag(pW1, nt*2 + 0, lane, b);
    c = wmma_bf16(a0, b, c);
    load_bfrag(pW1, nt*2 + 1, lane, b);
    c = wmma_bf16(a1, b, c);
    int col = nt*16 + lm;
    float bb = b1[col];
    #pragma unroll
    for (int r = 0; r < 8; ++r)
      hl[w][r + half*8][col] = ssilu(c[r] + bb);
  }
  __syncthreads();
  ABf a2;                                         // hidden 16x32 as A frag
  pack8(&hl[w][lm][     half*8], &a2.h4[0]);
  pack8(&hl[w][lm][16 + half*8], &a2.h4[2]);

  #pragma unroll
  for (int nt = 0; nt < 12; ++nt) {
    v8f c = {};
    ABf b;
    load_bfrag(pW2, nt, lane, b);
    c = wmma_bf16(a2, b, c);
    int col = nt*16 + lm;
    float bb = b2[col];
    #pragma unroll
    for (int r = 0; r < 8; ++r) {
      int rr = n0 + r + half*8;
      if (rr < N) sp[rr*C3H + col] = c[r] + bb;
    }
  }
}

// ---------- kernel 5: main edge kernel (1 wave = 16 edges) ----------
__global__ void __launch_bounds__(128) edge_main_k(
    const float* __restrict__ erbf, const float* __restrict__ udiff,
    const int* __restrict__ eidx,
    const unsigned* __restrict__ pWe, const float* __restrict__ be,
    const float* __restrict__ Winv, const float* __restrict__ binv,
    const float* __restrict__ lng, const float* __restrict__ lnb,
    const float* __restrict__ rvec, const float* __restrict__ rsh,
    const float* __restrict__ sp, const float* __restrict__ nvec,
    float* dsc, float* dvec, int N, int E) {
  __shared__ float geo[4][16][9];
  __shared__ float udl[4][16][3];
  __shared__ int   jil[4][16][2];
  int lane = threadIdx.x & 31, w = threadIdx.x >> 5;
  int half = lane >> 4, lm = lane & 15;
  int e0 = (blockIdx.x*4 + w) * 16;
  if (e0 >= E) return;                            // wave-uniform exit

  // --- phase 1: rbf_h = edge_rbf(16x64) @ W_edge(64x192), bf16 WMMA ---
  ABf a0, a1;
  {
    int er = e0 + lm; if (er >= E) er = E - 1;
    pack8(&erbf[er*R +      half*8], &a0.h4[0]);
    pack8(&erbf[er*R + 16 + half*8], &a0.h4[2]);
    pack8(&erbf[er*R + 32 + half*8], &a1.h4[0]);
    pack8(&erbf[er*R + 48 + half*8], &a1.h4[2]);
  }
  v8f acc[12];
  #pragma unroll
  for (int nt = 0; nt < 12; ++nt) {
    v8f c = {};
    ABf b;
    load_bfrag(pWe, nt*2 + 0, lane, b);
    c = wmma_bf16(a0, b, c);
    load_bfrag(pWe, nt*2 + 1, lane, b);
    c = wmma_bf16(a1, b, c);
    acc[nt] = c;
  }

  // --- phase 2: lanes 0..15 compute geometry (SH invariants -> LN -> geom) ---
  if (lane < 16) {
    int e = e0 + lane; if (e >= E) e = E - 1;
    float ux = udiff[e*3+0], uy = udiff[e*3+1], uz = udiff[e*3+2];
    int jj = eidx[e];
    int ii = eidx[E + e];
    jil[w][lane][0] = jj; jil[w][lane][1] = ii;
    udl[w][lane][0] = ux; udl[w][lane][1] = uy; udl[w][lane][2] = uz;
    float accl[NUM_INV] = {0.f, 0.f, 0.f, 0.f, 0.f, 0.f, 0.f};
    sph_dot7(ux, uy, uz, &rsh[ii*SH_DIM], accl);
    const float ILD[NUM_INV] = {1.0f, 1.0f/3.0f, 1.0f/5.0f, 1.0f/7.0f,
                                1.0f/9.0f, 1.0f/11.0f, 1.0f/13.0f};
    float inv[NUM_INV], mu = 0.0f;
    #pragma unroll
    for (int l = 0; l <= LMAX; ++l) { inv[l] = accl[l] * ILD[l]; mu += inv[l]; }
    mu *= (1.0f / 7.0f);
    float var = 0.0f;
    #pragma unroll
    for (int l = 0; l <= LMAX; ++l) { float d = inv[l] - mu; var += d*d; }
    var *= (1.0f / 7.0f);
    float rstd = rsqrtf(var + 1e-5f);
    float ct = ux*rvec[ii*3+0] + uy*rvec[ii*3+1] + uz*rvec[ii*3+2];
    geo[w][lane][0] = ct;
    #pragma unroll
    for (int l = 0; l <= LMAX; ++l)
      geo[w][lane][1 + l] = (inv[l] - mu) * rstd * lng[l] + lnb[l];
  }
  __syncthreads();

  // --- phase 3: gate, gather, scatter-add ---
  const float INV_SQRT3  = 0.57735026918962576f;
  const float INV_SQRT_H = 0.125f;                // 1/sqrt(64)
  #pragma unroll
  for (int n = 0; n < 4; ++n) {                   // x1 (cols h) & x2 (cols 64+h)
    int h = n*16 + lm;
    float w1[8], w2s[8];
    #pragma unroll
    for (int q = 0; q < 8; ++q) { w1[q] = Winv[q*C3H + h]; w2s[q] = Winv[q*C3H + 64 + h]; }
    float bi1 = binv[h], bi2 = binv[64 + h];
    float be1 = be[h],   be2 = be[64 + h];
    #pragma unroll
    for (int r = 0; r < 8; ++r) {
      int M = r + half*8;
      int e = e0 + M;
      if (e >= E) continue;
      float g1 = 0.0f, g2 = 0.0f;
      #pragma unroll
      for (int q = 0; q < 8; ++q) { float gq = geo[w][M][q]; g1 += gq*w1[q]; g2 += gq*w2s[q]; }
      g1 = sgate(g1 + bi1); g2 = sgate(g2 + bi2);
      int jj = jil[w][M][0], ii = jil[w][M][1];
      float r1 = (acc[n][r]     + be1) * (1.0f + g1);
      float r2 = (acc[n + 4][r] + be2) * (1.0f + g2);
      float x1v = sp[jj*C3H + h]      * r1 * INV_SQRT3;
      float x2v = sp[jj*C3H + 64 + h] * r2 * INV_SQRT3;
      #pragma unroll
      for (int d = 0; d < 3; ++d) {
        float vv = (x1v * nvec[jj*C3H + d*H + h] + x2v * udl[w][M][d]) * INV_SQRT_H;
        unsafeAtomicAdd(&dvec[ii*C3H + d*H + h], vv);
      }
    }
  }
  #pragma unroll
  for (int n = 8; n < 12; ++n) {                  // x3 (cols 128+h) -> delta_scalar
    int h = (n - 8)*16 + lm;
    int col = 128 + h;
    float w3[8];
    #pragma unroll
    for (int q = 0; q < 8; ++q) w3[q] = Winv[q*C3H + col];
    float bi3 = binv[col], be3 = be[col];
    #pragma unroll
    for (int r = 0; r < 8; ++r) {
      int M = r + half*8;
      int e = e0 + M;
      if (e >= E) continue;
      float g3 = 0.0f;
      #pragma unroll
      for (int q = 0; q < 8; ++q) g3 += geo[w][M][q] * w3[q];
      g3 = sgate(g3 + bi3);
      int jj = jil[w][M][0], ii = jil[w][M][1];
      float r3  = (acc[n][r] + be3) * (1.0f + g3);
      float x3v = sp[jj*C3H + col] * r3 * INV_SQRT3;
      unsafeAtomicAdd(&dsc[ii*H + h], x3v);
    }
  }
}

// ---------- host launcher ----------
extern "C" void kernel_launch(void* const* d_in, const int* in_sizes, int n_in,
                              void* d_out, int out_size, void* d_ws, size_t ws_size,
                              hipStream_t stream) {
  (void)n_in; (void)ws_size;
  const float* node_scalar = (const float*)d_in[0];
  const float* node_vector = (const float*)d_in[1];
  const float* edge_rbf    = (const float*)d_in[2];
  const float* edge_udiff  = (const float*)d_in[3];
  const int*   edge_index  = (const int*)d_in[4];
  const float* W_edge = (const float*)d_in[5];
  const float* b_edge = (const float*)d_in[6];
  const float* W_x1   = (const float*)d_in[7];
  const float* b_x1   = (const float*)d_in[8];
  const float* W_x2   = (const float*)d_in[9];
  const float* b_x2   = (const float*)d_in[10];
  const float* ln_g   = (const float*)d_in[11];
  const float* ln_b   = (const float*)d_in[12];
  const float* W_inv  = (const float*)d_in[13];
  const float* b_inv  = (const float*)d_in[14];

  int N = in_sizes[0] / H;
  int E = in_sizes[3] / 3;

  char* ws = (char*)d_ws;
  size_t off = 0;
  auto alloc = [&](size_t bytes) -> char* {
    char* p = ws + off;
    off += (bytes + 255) & ~(size_t)255;
    return p;
  };
  unsigned* pWe = (unsigned*)alloc(6144 * 4);
  unsigned* pW1 = (unsigned*)alloc(1024 * 4);
  unsigned* pW2 = (unsigned*)alloc(3072 * 4);
  float* cnt  = (float*)alloc((size_t)N * 4);
  float* rvec = (float*)alloc((size_t)N * 3 * 4);
  float* rsh  = (float*)alloc((size_t)N * SH_DIM * 4);
  float* sprj = (float*)alloc((size_t)N * C3H * 4);

  float* dsc  = (float*)d_out;                   // delta_scalar (N*H)
  float* dvec = dsc + (size_t)N * H;             // delta_vector (N*3*H)

  hipMemsetAsync(cnt,  0, (size_t)N * 4, stream);
  hipMemsetAsync(rvec, 0, (size_t)N * 3 * 4, stream);
  hipMemsetAsync(d_out, 0, (size_t)out_size * 4, stream);

  pack_weights<<<40, 256, 0, stream>>>(W_edge, W_x1, W_x2, pWe, pW1, pW2);
  edge_ref_scatter<<<(E + 255) / 256, 256, 0, stream>>>(edge_udiff, edge_index, E, rvec, cnt);
  node_ref<<<(N + 255) / 256, 256, 0, stream>>>(rvec, cnt, rsh, N);
  scalar_proj_k<<<(N + 63) / 64, 128, 0, stream>>>(node_scalar, pW1, pW2, b_x1, b_x2, sprj, N);
  edge_main_k<<<(E + 63) / 64, 128, 0, stream>>>(edge_rbf, edge_udiff, edge_index,
      pWe, b_edge, W_inv, b_inv, ln_g, ln_b, rvec, rsh, sprj, node_vector,
      dsc, dvec, N, E);
}
